// DiscriminatorNetwork_23106924052892
// MI455X (gfx1250) — matrix operand order repaired and checked by reading, not compiled, not run on hardware
//
#include <hip/hip_runtime.h>
#include <hip/hip_bf16.h>
#include <math.h>

#define HH 512
#define BB 256
#define TT 512
#define NBLOCKS 64
#define STRIDE_DW 520   // 512 dwords per 1024-col bf16 row + 8 dwords pad (bank-conflict avoidance)

typedef __attribute__((ext_vector_type(16))) __bf16 v16bf;
typedef __attribute__((ext_vector_type(8)))  float  v8f;
typedef __attribute__((ext_vector_type(4)))  unsigned int u32x4;   // raw 128-bit vector (no HIP class)

// ---------------- workspace layout (bytes) ----------------
#define SYNC_OFF   0u
#define PW0_OFF    1024u
#define PW0_BYTES  (128u*16u*1024u)              // layer0: 128 ntiles x 16 ktiles x 1KB frag
#define PW1_OFF    (PW0_OFF + PW0_BYTES)
#define PW1_BYTES  (128u*32u*1024u)              // layer1: K=1024 -> 32 ktiles
#define PW2_OFF    (PW1_OFF + PW1_BYTES)
#define PW2_BYTES  (128u*32u*1024u)
#define BIAS_OFF   (PW2_OFF + PW2_BYTES)
#define BIAS_BYTES (3u*2048u*4u)
#define HBUF_OFF   (BIAS_OFF + BIAS_BYTES)       // 3 layers x 2 (double buffer) x [256x512] bf16
#define HBUF_BYTES (3u*2u*(unsigned)BB*(unsigned)HH*2u)
#define H2F_OFF    (HBUF_OFF + HBUF_BYTES)       // final layer-2 h in f32
#define H2F_BYTES  ((unsigned)BB*(unsigned)HH*4u)
#define XT_OFF     (H2F_OFF + H2F_BYTES)         // tracks transposed to [T][2][B] f32
#define XT_BYTES   ((unsigned)TT*2u*(unsigned)BB*4u)

// -------- explicit global (address_space(1)) memory helpers: force global_load/store --------
#define GAS __attribute__((address_space(1)))
__device__ __forceinline__ u32x4 gload16(const void* p) {
  return *(const GAS u32x4*)(unsigned long long)p;
}
__device__ __forceinline__ float gloadf(const void* p) {
  return *(const GAS float*)(unsigned long long)p;
}
__device__ __forceinline__ void gstore16(void* p, u32x4 v) {
  *(GAS u32x4*)(unsigned long long)p = v;
}
__device__ __forceinline__ void gstore_u16(void* p, unsigned short v) {
  *(GAS unsigned short*)(unsigned long long)p = v;
}
__device__ __forceinline__ void gstoref(void* p, float v) {
  *(GAS float*)(unsigned long long)p = v;
}

__device__ __forceinline__ unsigned short f2bf(float f) {
  unsigned u = __builtin_bit_cast(unsigned, f);
  u += 0x7fffu + ((u >> 16) & 1u);               // round-to-nearest-even
  return (unsigned short)(u >> 16);
}

// branch-free transcendentals: v_exp_f32 + v_rcp_f32
__device__ __forceinline__ float sigm(float x) {
  return __builtin_amdgcn_rcpf(1.0f + __expf(-x));
}
__device__ __forceinline__ float tanh_fast(float x) {
  float e = __expf(-2.0f * x);
  return (1.0f - e) * __builtin_amdgcn_rcpf(1.0f + e);
}

// ---------------- prep: fused bias per layer ----------------
__global__ void prep_bias(const float* bih0, const float* bhh0,
                          const float* bih1, const float* bhh1,
                          const float* bih2, const float* bhh2,
                          float* bias) {
  int id = blockIdx.x * 256 + threadIdx.x;
  if (id >= 3 * 2048) return;
  int l = id >> 11, n = id & 2047;
  const float* a = (l == 0) ? bih0 : ((l == 1) ? bih1 : bih2);
  const float* b = (l == 0) ? bhh0 : ((l == 1) ? bhh1 : bhh2);
  gstoref(bias + id, gloadf(a + n) + gloadf(b + n));
}

// ---------------- prep: transpose tracks [B,T,2] -> [T][2][B] ----------------
__global__ void prep_tracks(const float* __restrict__ tracks, float* __restrict__ xt) {
  int id = blockIdx.x * 256 + threadIdx.x;       // B*T*2 = 262144
  if (id >= BB * TT * 2) return;
  int c = id & 1;
  int t = (id >> 1) % TT;
  int b = (id >> 1) / TT;
  gstoref(xt + (t * 2 + c) * BB + b, gloadf(tracks + id));
}

// ---------------- prep: pack weights into bf16 B-fragment layout ----------------
// B-frag (32x16 bf16, KxN), wave32: VGPR v, lane L holds pair {B[kb,n], B[kb+1,n]}
// with kb = kt*32 + 2v + (L>=16 ? 16 : 0), n = nbase + (L&15).  B[k,n] = W[n,k].
// Packed as: pw[((ntile*nK + kt)*32 + lane)*8 + v]  (dwords) -> main loop loads 2x b128/lane.
__global__ void pack_weights(const float* __restrict__ Whh0,
                             const float* __restrict__ Wih1, const float* __restrict__ Whh1,
                             const float* __restrict__ Wih2, const float* __restrict__ Whh2,
                             unsigned char* __restrict__ ws) {
  int id = blockIdx.x * 256 + threadIdx.x;
  int l, rem, nK;
  unsigned pwoff;
  const int L0 = 128 * 16 * 32, L12 = 128 * 32 * 32;
  if (id < L0)            { l = 0; rem = id;            nK = 16; pwoff = PW0_OFF; }
  else if (id < L0 + L12) { l = 1; rem = id - L0;       nK = 32; pwoff = PW1_OFF; }
  else                    { l = 2; rem = id - L0 - L12; nK = 32; pwoff = PW2_OFF; }
  int lane = rem & 31;
  int kt   = (rem >> 5) % nK;
  int nt   = (rem >> 5) / nK;
  int n     = nt * 16 + (lane & 15);
  int kbase = kt * 32 + ((lane >> 4) << 4);
  const float* Wi = (l == 1) ? Wih1 : Wih2;
  const float* Wh = (l == 0) ? Whh0 : ((l == 1) ? Whh1 : Whh2);
  unsigned int d[8];
#pragma unroll
  for (int v = 0; v < 8; ++v) {
    int k = kbase + 2 * v;
    float a, b;
    if (l == 0)      { a = gloadf(Wh + n * HH + k);        b = gloadf(Wh + n * HH + k + 1); }
    else if (k < HH) { a = gloadf(Wi + n * HH + k);        b = gloadf(Wi + n * HH + k + 1); }
    else             { a = gloadf(Wh + n * HH + (k - HH)); b = gloadf(Wh + n * HH + (k - HH) + 1); }
    d[v] = (unsigned)f2bf(a) | ((unsigned)f2bf(b) << 16);
  }
  unsigned char* dst = ws + pwoff + ((((unsigned)(nt * nK + kt)) << 8) + lane * 8u) * 4u;
  gstore16(dst,      *(const u32x4*)&d[0]);
  gstore16(dst + 16, *(const u32x4*)&d[4]);
}

// ---------------- grid-wide barrier (monotonic counter, reset per launch) ----------------
__device__ __forceinline__ void gsync(unsigned int* ctr, unsigned int& gen) {
  __syncthreads();
  if (threadIdx.x == 0) {
    gen += 1;
    __hip_atomic_fetch_add(ctr, 1u, __ATOMIC_RELEASE, __HIP_MEMORY_SCOPE_AGENT);
    while (__hip_atomic_load(ctr, __ATOMIC_ACQUIRE, __HIP_MEMORY_SCOPE_AGENT) <
           gen * (unsigned)NBLOCKS) {
      __builtin_amdgcn_s_sleep(2);
    }
  }
  __syncthreads();
}

// stage 16 rows x 512 bf16 of h (row-major, stride HH) into LDS half {0,1}
__device__ __forceinline__ void stageA(unsigned int* lds, const unsigned short* src, int half) {
  int tid = threadIdx.x;
#pragma unroll
  for (int u = 0; u < 4; ++u) {
    int g = tid + u * 256;          // 128-bit chunk index, 0..1023 (16 rows x 64 chunks)
    int row = g >> 6, c = g & 63;
    u32x4 d = gload16(src + row * HH + c * 8);
    *(u32x4*)(lds + row * STRIDE_DW + half * 256 + c * 4) = d;
  }
}

// ---------------- persistent fused LSTM ----------------
__launch_bounds__(256, 1)
__global__ void lstm_persist(const float* __restrict__ Wih0,
                             const float* __restrict__ Wpred,
                             const float* __restrict__ bpred,
                             unsigned char* __restrict__ ws,
                             float* __restrict__ out) {
  __shared__ __align__(16) unsigned int lds[16 * STRIDE_DW];

  const int tid   = threadIdx.x;
  const int wave  = tid >> 5;
  const int lane  = tid & 31;
  const int mtile = blockIdx.x >> 2;            // 0..15  (batch rows)
  const int ngrp  = blockIdx.x & 3;
  const int ntile = ngrp * 8 + wave;            // 0..31  (hidden cols)
  const int m0    = mtile * 16;
  const int n0    = ntile * 16;
  const int lnlo  = lane & 15;
  const int hi    = lane >> 4;

  unsigned int*   syncc = (unsigned int*)(ws + SYNC_OFF);
  const float*    bias  = (const float*)(ws + BIAS_OFF);
  unsigned short* hbuf  = (unsigned short*)(ws + HBUF_OFF);
  float*          h2f   = (float*)(ws + H2F_OFF);
  const float*    xt    = (const float*)(ws + XT_OFF);

  // per-gate constants (column = n0 + lnlo + g*H)
  float biasv[3][4], wia[4], wib[4];
#pragma unroll
  for (int g = 0; g < 4; ++g) {
    int ng = n0 + lnlo + g * HH;
    biasv[0][g] = gloadf(bias + ng);
    biasv[1][g] = gloadf(bias + 2048 + ng);
    biasv[2][g] = gloadf(bias + 4096 + ng);
    wia[g] = gloadf(Wih0 + ng * 2);
    wib[g] = gloadf(Wih0 + ng * 2 + 1);
  }

  const v8f vzero = {0.f, 0.f, 0.f, 0.f, 0.f, 0.f, 0.f, 0.f};
  v8f cfr[3] = {vzero, vzero, vzero};          // cell state lives in registers all T steps
  unsigned gen = 0;

  union BF { unsigned int d[8]; v16bf v; };

  for (int t = 0; t < TT; ++t) {
    const int wb = t & 1, rb = wb ^ 1;
    for (int l = 0; l < 3; ++l) {
      // ---- stage A tile(s) into LDS, shared by the 8 waves of this block ----
      int nK;
      if (l == 0) {
        stageA(lds, hbuf + (unsigned)(0 * 2 + rb) * (BB * HH) + m0 * HH, 0);
        nK = 16;                                  // K = 512 (recurrent only)
      } else {
        stageA(lds, hbuf + (unsigned)((l - 1) * 2 + wb) * (BB * HH) + m0 * HH, 0); // h_below(t)
        stageA(lds, hbuf + (unsigned)(l * 2 + rb) * (BB * HH) + m0 * HH, 1);       // h_self(t-1)
        nK = 32;                                  // K = 1024 fused
      }
      __syncthreads();

      // ---- gate GEMM: 4 gates as 4 WMMA accumulators ----
      v8f acc[4];
#pragma unroll
      for (int g = 0; g < 4; ++g) {
        float bv = biasv[l][g];
        acc[g] = (v8f){bv, bv, bv, bv, bv, bv, bv, bv};
      }
      // per-layer packed-weight base + per-gate fragment byte bases (hoisted)
      const unsigned pwoff = (l == 0) ? PW0_OFF : ((l == 1) ? PW1_OFF : PW2_OFF);
      const unsigned char* pwl = ws + pwoff;
      unsigned bgo[4];
#pragma unroll
      for (int g = 0; g < 4; ++g)
        bgo[g] = ((unsigned)(ntile + g * 32) * (unsigned)nK) * 1024u + (unsigned)lane * 32u;

      // register double-buffered software pipeline (distance 2 on B, 2 on A)
      BF Bb[2][4];
      BF Ab[2];
      auto loadB = [&](int buf, int kt) {
#pragma unroll
        for (int g = 0; g < 4; ++g) {
          const unsigned char* bp = pwl + (bgo[g] + (unsigned)kt * 1024u);
          *(u32x4*)&Bb[buf][g].d[0] = gload16(bp);
          *(u32x4*)&Bb[buf][g].d[4] = gload16(bp + 16);
        }
      };
      auto loadA = [&](int buf, int kt) {
        // A frag 16x32 bf16: lane<16 holds K={k0..k0+15}; lane>=16 holds K={k0+16..k0+31}
        const unsigned int* ap = lds + lnlo * STRIDE_DW + kt * 16 + hi * 8;
        *(u32x4*)&Ab[buf].d[0] = *(const u32x4*)ap;
        *(u32x4*)&Ab[buf].d[4] = *(const u32x4*)(ap + 4);
      };
      loadB(0, 0); loadA(0, 0);
      loadB(1, 1); loadA(1, 1);

#pragma unroll 2
      for (int kt = 0; kt < nK; ++kt) {
        const int cur = kt & 1;
#pragma unroll
        for (int g = 0; g < 4; ++g) {
          acc[g] = __builtin_amdgcn_wmma_f32_16x16x32_bf16(
              false, Ab[cur].v, false, Bb[cur][g].v, (short)0, acc[g], false, false);
        }
        const int kt2 = (kt + 2 < nK) ? kt + 2 : kt;   // branchless clamp (re-load, harmless)
        loadB(cur, kt2);
        loadA(cur, kt2);
      }

      // ---- layer 0: 2-wide input term; xt[t][c][b] -> uniform per half-wave (broadcast) ----
      if (l == 0) {
        const float* xtp = xt + t * 2 * BB;
#pragma unroll
        for (int r = 0; r < 8; ++r) {
          int brow = m0 + r + hi * 8;
          float x0 = gloadf(xtp + brow);
          float x1 = gloadf(xtp + BB + brow);
          acc[0][r] += x0 * wia[0] + x1 * wib[0];
          acc[1][r] += x0 * wia[1] + x1 * wib[1];
          acc[2][r] += x0 * wia[2] + x1 * wib[2];
          acc[3][r] += x0 * wia[3] + x1 * wib[3];
        }
      }

      // ---- LSTM pointwise on fragments; write h (bf16) ----
      unsigned short* hout = hbuf + (unsigned)(l * 2 + wb) * (BB * HH);
#pragma unroll
      for (int r = 0; r < 8; ++r) {
        float iv = sigm(acc[0][r]);
        float fv = sigm(acc[1][r]);
        float gv = tanh_fast(acc[2][r]);
        float ov = sigm(acc[3][r]);
        float c  = fv * cfr[l][r] + iv * gv;
        cfr[l][r] = c;
        float h = ov * tanh_fast(c);
        int row = m0 + r + hi * 8;
        gstore_u16(hout + row * HH + n0 + lnlo, f2bf(h));
        if (l == 2 && t == TT - 1) gstoref(h2f + row * HH + n0 + lnlo, h);
      }
      gsync(syncc, gen);
    }
  }

  // ---- epilogue: ELU + [256x512]x[512x4] projection (tiny) ----
  if (blockIdx.x == 0) {
    int b = tid;  // 256 threads, one batch row each
    float a0 = gloadf(bpred + 0), a1 = gloadf(bpred + 1);
    float a2 = gloadf(bpred + 2), a3 = gloadf(bpred + 3);
    for (int j = 0; j < HH; ++j) {
      float v = gloadf(h2f + b * HH + j);
      v = (v > 0.f) ? v : (__expf(v) - 1.f);
      a0 += v * gloadf(Wpred + j);
      a1 += v * gloadf(Wpred + HH + j);
      a2 += v * gloadf(Wpred + 2 * HH + j);
      a3 += v * gloadf(Wpred + 3 * HH + j);
    }
    gstoref(out + b * 4 + 0, a0); gstoref(out + b * 4 + 1, a1);
    gstoref(out + b * 4 + 2, a2); gstoref(out + b * 4 + 3, a3);
  }
}

extern "C" void kernel_launch(void* const* d_in, const int* in_sizes, int n_in,
                              void* d_out, int out_size, void* d_ws, size_t ws_size,
                              hipStream_t stream) {
  (void)in_sizes; (void)n_in; (void)out_size; (void)ws_size;
  const float* tracks = (const float*)d_in[0];
  const float* Wih0 = (const float*)d_in[1];
  const float* bih0 = (const float*)d_in[2];
  const float* Whh0 = (const float*)d_in[3];
  const float* bhh0 = (const float*)d_in[4];
  const float* Wih1 = (const float*)d_in[5];
  const float* bih1 = (const float*)d_in[6];
  const float* Whh1 = (const float*)d_in[7];
  const float* bhh1 = (const float*)d_in[8];
  const float* Wih2 = (const float*)d_in[9];
  const float* bih2 = (const float*)d_in[10];
  const float* Whh2 = (const float*)d_in[11];
  const float* bhh2 = (const float*)d_in[12];
  const float* Wpred = (const float*)d_in[13];
  const float* bpred = (const float*)d_in[14];
  unsigned char* ws = (unsigned char*)d_ws;

  // deterministic per-launch reset: barrier counter + zero-initial hidden states
  (void)hipMemsetAsync(ws + SYNC_OFF, 0, 256, stream);
  (void)hipMemsetAsync(ws + HBUF_OFF, 0, HBUF_BYTES, stream);

  prep_bias<<<24, 256, 0, stream>>>(bih0, bhh0, bih1, bhh1, bih2, bhh2,
                                    (float*)(ws + BIAS_OFF));
  prep_tracks<<<1024, 256, 0, stream>>>(tracks, (float*)(ws + XT_OFF));
  pack_weights<<<1280, 256, 0, stream>>>(Whh0, Wih1, Whh1, Wih2, Whh2, ws);
  lstm_persist<<<NBLOCKS, 256, 0, stream>>>(Wih0, Wpred, bpred, ws, (float*)d_out);
}
